// ROIHead_37314675868403
// MI455X (gfx1250) — matrix operand order repaired and checked by reading, not compiled, not run on hardware
//
#include <hip/hip_runtime.h>
#include <hip/hip_bf16.h>

// ---------------- problem constants (match reference) ----------------
#define NPROP     1000
#define MPAD      1024            // 1000 padded to multiple of 32 (macro-tile rows)
#define CCH       512
#define HF        50
#define WF        50
#define PP        7
#define DK        25088           // 512*7*7, multiple of 32
#define F6        1024
#define NCLS      52
#define NCLS_PAD  64
#define NBOX      208             // 52*4, multiple of 16
#define NEL       51000           // 1000*51 candidates
#define TOPK      2000
#define MAXDET    100
#define SCALE_RP  (1.0f/16.0f)
#define SCORE_THR 0.05f
#define NMS_IOU_T 0.5f

typedef __attribute__((ext_vector_type(16))) __bf16 v16bf;
typedef __attribute__((ext_vector_type(8)))  float  v8f;

union ABFrag { v16bf v; uint4 u[2]; };

// ---------------- 1. fp32 -> bf16 convert with row padding ----------------
__global__ void cvt_pad_bf16(const float* __restrict__ src, __bf16* __restrict__ dst,
                             int rows, int cols, int prows) {
    long long idx = (long long)blockIdx.x * blockDim.x + threadIdx.x;
    long long total = (long long)prows * cols;
    if (idx >= total) return;
    int r = (int)(idx / cols);
    float v = (r < rows) ? src[idx] : 0.0f;
    dst[idx] = (__bf16)v;
}

// ---------------- 2. ROI max pool -> bf16 A matrix [MPAD x DK] ----------------
__global__ void roi_pool_kernel(const float* __restrict__ feat,
                                const float* __restrict__ props,
                                __bf16* __restrict__ A) {
    int n = blockIdx.x;
    if (n >= NPROP) {          // zero pad rows (block-uniform branch)
        for (int t = threadIdx.x; t < DK; t += blockDim.x)
            A[(size_t)n * DK + t] = (__bf16)0.0f;
        return;
    }
    float x1 = props[n*4+0], y1 = props[n*4+1], x2 = props[n*4+2], y2 = props[n*4+3];
    int rs_w = (int)floorf(x1 * SCALE_RP + 0.5f);
    int rs_h = (int)floorf(y1 * SCALE_RP + 0.5f);
    int re_w = (int)floorf(x2 * SCALE_RP + 0.5f);
    int re_h = (int)floorf(y2 * SCALE_RP + 0.5f);
    float roi_w = fmaxf((float)(re_w - rs_w + 1), 1.0f);
    float roi_h = fmaxf((float)(re_h - rs_h + 1), 1.0f);
    float bin_w = roi_w / (float)PP, bin_h = roi_h / (float)PP;

    for (int idx = threadIdx.x; idx < DK; idx += blockDim.x) {
        int c  = idx / (PP*PP);
        int b  = idx % (PP*PP);
        int ph = b / PP, pw = b % PP;
        int hs = min(max((int)floorf((float)ph * bin_h)       + rs_h, 0), HF);
        int he = min(max((int)ceilf ((float)(ph+1) * bin_h)   + rs_h, 0), HF);
        int ws = min(max((int)floorf((float)pw * bin_w)       + rs_w, 0), WF);
        int we = min(max((int)ceilf ((float)(pw+1) * bin_w)   + rs_w, 0), WF);
        float m = 0.0f;
        if (he > hs && we > ws) {
            m = -3.0e38f;
            const float* fc = feat + (size_t)c * (HF*WF);
            for (int h = hs; h < he; ++h)
                for (int w = ws; w < we; ++w)
                    m = fmaxf(m, fc[h*WF + w]);
        }
        A[(size_t)n * DK + idx] = (__bf16)m;
    }
}

// ---------------- 3. bf16 WMMA GEMM with register blocking ----------------
// A: [M x K] bf16 row-major (lda=K), W: [N x K] bf16 row-major (ldw=K).
// One wave computes a (16*BM) x (16*BN) macro-tile: BM*BN accumulators fed by
// BM A-fragments and BN B-fragments per K=32 step (reuse cuts L2 traffic).
template<int BM, int BN>
__global__ void gemm_wmma_bf16(const __bf16* __restrict__ A, int lda,
                               const __bf16* __restrict__ W, int ldw,
                               const float* __restrict__ bias, int nbias,
                               float* __restrict__ Cf, __bf16* __restrict__ Cb,
                               int ldc, int M, int N, int K, int relu) {
    int wave   = (blockIdx.x * blockDim.x + threadIdx.x) >> 5;
    int lane   = threadIdx.x & 31;
    int tilesN = N / (16 * BN);
    int tiles  = (M / (16 * BM)) * tilesN;
    if (wave >= tiles) return;                 // wave-uniform: EXEC stays all-ones
    int tm = (wave / tilesN) * (16 * BM);
    int tn = (wave % tilesN) * (16 * BN);
    int r  = lane & 15;
    int hi = lane >> 4;                        // 0: low K-half lanes, 1: high

    // ISA 7.12.2: 16-bit A 16x32 — lane<16 holds K 0..7 & 16..23 of row r,
// lane>=16 holds K 8..15 & 24..31.  B 32x16 — lane maps to column N=r,
    // same K order as A: lanes<16 hold K 0..7 & 16..23, lanes>=16 hold K 8..15 & 24..31.
    const __bf16* arow[BM];
    const __bf16* brow[BN];
#pragma unroll
    for (int i = 0; i < BM; ++i) arow[i] = A + (size_t)(tm + i*16 + r) * lda + hi * 8;
#pragma unroll
    for (int j = 0; j < BN; ++j) brow[j] = W + (size_t)(tn + j*16 + r) * ldw + hi * 8;

    v8f acc[BM][BN] = {};
    for (int k = 0; k < K; k += 32) {
        ABFrag a[BM], b[BN];
#pragma unroll
        for (int i = 0; i < BM; ++i) {
            a[i].u[0] = *(const uint4*)(arow[i] + k);
            a[i].u[1] = *(const uint4*)(arow[i] + k + 16);
            __builtin_prefetch((const void*)(arow[i] + k + 512), 0, 1);
        }
#pragma unroll
        for (int j = 0; j < BN; ++j) {
            b[j].u[0] = *(const uint4*)(brow[j] + k);
            b[j].u[1] = *(const uint4*)(brow[j] + k + 16);
            __builtin_prefetch((const void*)(brow[j] + k + 512), 0, 1);
        }
#pragma unroll
        for (int i = 0; i < BM; ++i)
#pragma unroll
            for (int j = 0; j < BN; ++j)
                acc[i][j] = __builtin_amdgcn_wmma_f32_16x16x32_bf16(
                    false, a[i].v, false, b[j].v, (short)0, acc[i][j], false, false);
    }

    // C/D layout per tile: col = tn' + (lane&15); VGPR v -> row = tm' + hi*8 + v
#pragma unroll
    for (int j = 0; j < BN; ++j) {
        int col = tn + j * 16 + r;
        float bv = (bias != nullptr && col < nbias) ? bias[col] : 0.0f;
#pragma unroll
        for (int i = 0; i < BM; ++i) {
#pragma unroll
            for (int v = 0; v < 8; ++v) {
                int row = tm + i * 16 + hi * 8 + v;
                float x = acc[i][j][v] + bv;
                if (relu) x = fmaxf(x, 0.0f);
                if (Cf) Cf[(size_t)row * ldc + col] = x;
                if (Cb) Cb[(size_t)row * ldc + col] = (__bf16)x;
            }
        }
    }
}

// ---------------- 4. softmax + box decode + validity ----------------
__global__ void decode_kernel(const float* __restrict__ cls,   // [MPAD x 64]
                              const float* __restrict__ bbx,   // [MPAD x 208]
                              const float* __restrict__ props,
                              const int* __restrict__ imh, const int* __restrict__ imw,
                              float* __restrict__ boxes,       // [NEL x 4]
                              float* __restrict__ raw,         // [NEL]
                              float* __restrict__ masked,      // [NEL]
                              int*   __restrict__ valid) {     // [NEL]
    int n = blockIdx.x * blockDim.x + threadIdx.x;
    if (n >= NPROP) return;
    const float* cr = cls + (size_t)n * NCLS_PAD;
    float mx = -3.0e38f;
    for (int j = 0; j < NCLS; ++j) mx = fmaxf(mx, cr[j]);
    float sum = 0.0f;
    for (int j = 0; j < NCLS; ++j) sum += __expf(cr[j] - mx);
    float inv = 1.0f / sum;

    float x1 = props[n*4+0], y1 = props[n*4+1], x2 = props[n*4+2], y2 = props[n*4+3];
    float pw = x2 - x1, ph = y2 - y1;
    float pcx = x1 + 0.5f * pw, pcy = y1 + 0.5f * ph;
    float limw = (float)imw[0], limh = (float)imh[0];

    const float* tr = bbx + (size_t)n * NBOX;
    for (int j = 1; j < NCLS; ++j) {
        float t0 = tr[j*4+0], t1 = tr[j*4+1], t2 = tr[j*4+2], t3 = tr[j*4+3];
        float gcx = t0 * pw + pcx, gcy = t1 * ph + pcy;
        float gw  = __expf(t2) * pw, gh = __expf(t3) * ph;
        float bx1 = fminf(fmaxf(gcx - 0.5f * gw, 0.0f), limw);
        float by1 = fminf(fmaxf(gcy - 0.5f * gh, 0.0f), limh);
        float bx2 = fminf(fmaxf(gcx + 0.5f * gw, 0.0f), limw);
        float by2 = fminf(fmaxf(gcy + 0.5f * gh, 0.0f), limh);
        float sc  = __expf(cr[j] - mx) * inv;
        int  ok   = (sc > SCORE_THR) && ((bx2 - bx1) >= 1.0f) && ((by2 - by1) >= 1.0f);
        int  e    = n * (NCLS - 1) + (j - 1);
        boxes[e*4+0] = bx1; boxes[e*4+1] = by1; boxes[e*4+2] = bx2; boxes[e*4+3] = by2;
        raw[e]    = sc;
        masked[e] = ok ? sc : -1.0f;
        valid[e]  = ok;
    }
}

// ---------------- 5. exact stable top-K by rank counting ----------------
#define RCHUNK 2048
__global__ void topk_rank_kernel(const float* __restrict__ masked,
                                 const float* __restrict__ raw,
                                 const float* __restrict__ boxes,
                                 const int* __restrict__ valid,
                                 float* __restrict__ top_b, float* __restrict__ top_s,
                                 int* __restrict__ top_l, int* __restrict__ top_v) {
    __shared__ float sm[RCHUNK];
    int i = blockIdx.x * blockDim.x + threadIdx.x;
    float my = (i < NEL) ? masked[i] : 0.0f;
    int rank = 0;
    for (int base = 0; base < NEL; base += RCHUNK) {
        int cnt = min(RCHUNK, NEL - base);
        for (int t = threadIdx.x; t < cnt; t += blockDim.x) sm[t] = masked[base + t];
        __syncthreads();
        if (i < NEL) {
            for (int t = 0; t < cnt; ++t) {
                float s = sm[t];
                int j = base + t;
                rank += (s > my) || ((s == my) && (j < i));   // stable top_k tie-break
            }
        }
        __syncthreads();
    }
    if (i < NEL && rank < TOPK) {
        top_s[rank] = raw[i];
        top_b[rank*4+0] = boxes[i*4+0];
        top_b[rank*4+1] = boxes[i*4+1];
        top_b[rank*4+2] = boxes[i*4+2];
        top_b[rank*4+3] = boxes[i*4+3];
        top_l[rank] = (i % (NCLS - 1)) + 1;
        top_v[rank] = valid[i];
    }
}

// ---------------- 6. greedy NMS (class-offset boxes in LDS) + scatter ----------------
__global__ void nms_scatter_kernel(const float* __restrict__ top_b,
                                   const float* __restrict__ top_s,
                                   const int* __restrict__ top_l,
                                   const int* __restrict__ top_v,
                                   float* __restrict__ out) {
    __shared__ float bx[TOPK][4];     // 32 KB
    __shared__ float area[TOPK];      //  8 KB
    __shared__ unsigned char keep[TOPK];
    int tid = threadIdx.x;
    for (int t = tid; t < TOPK; t += blockDim.x) {
        float off = (float)top_l[t] * 1000.0f;
        float a = top_b[t*4+0] + off, b = top_b[t*4+1] + off;
        float c = top_b[t*4+2] + off, d = top_b[t*4+3] + off;
        bx[t][0] = a; bx[t][1] = b; bx[t][2] = c; bx[t][3] = d;
        area[t]  = (c - a) * (d - b);
        keep[t]  = (unsigned char)(top_v[t] != 0);
    }
    if (tid < (MAXDET*4 + MAXDET + MAXDET)) out[tid] = 0.0f;   // zero d_out (600 floats)
    __syncthreads();

    for (int i = 0; i < TOPK; ++i) {
        if (keep[i]) {                        // uniform read of LDS
            float ax1 = bx[i][0], ay1 = bx[i][1], ax2 = bx[i][2], ay2 = bx[i][3];
            float aa = area[i];
            for (int j = i + 1 + tid; j < TOPK; j += blockDim.x) {
                if (!keep[j]) continue;
                float xl = fmaxf(ax1, bx[j][0]);
                float yt = fmaxf(ay1, bx[j][1]);
                float xr = fminf(ax2, bx[j][2]);
                float yb = fminf(ay2, bx[j][3]);
                float inter = fmaxf(xr - xl, 0.0f) * fmaxf(yb - yt, 0.0f);
                float iou = inter / (aa + area[j] - inter);
                if (iou > NMS_IOU_T) keep[j] = 0;
            }
        }
        __syncthreads();
    }

    if (tid == 0) {
        int rank = 0;
        for (int i = 0; i < TOPK; ++i) {
            if (!keep[i]) continue;
            if (rank < MAXDET) {
                out[rank*4+0] = top_b[i*4+0];
                out[rank*4+1] = top_b[i*4+1];
                out[rank*4+2] = top_b[i*4+2];
                out[rank*4+3] = top_b[i*4+3];
                out[MAXDET*4 + rank]          = (float)top_l[i];   // labels
                out[MAXDET*4 + MAXDET + rank] = top_s[i];          // scores
            }
            ++rank;
        }
    }
}

// ---------------- launcher ----------------
static inline size_t align256(size_t x) { return (x + 255) & ~(size_t)255; }

extern "C" void kernel_launch(void* const* d_in, const int* in_sizes, int n_in,
                              void* d_out, int out_size, void* d_ws, size_t ws_size,
                              hipStream_t stream) {
    (void)in_sizes; (void)n_in; (void)out_size; (void)ws_size;
    const float* feat  = (const float*)d_in[0];
    const float* props = (const float*)d_in[1];
    const float* W6    = (const float*)d_in[2];
    const float* b6    = (const float*)d_in[3];
    const float* W7    = (const float*)d_in[4];
    const float* b7    = (const float*)d_in[5];
    const float* Wc    = (const float*)d_in[6];
    const float* bc    = (const float*)d_in[7];
    const float* Wb    = (const float*)d_in[8];
    const float* bb    = (const float*)d_in[9];
    const int*   imh   = (const int*)d_in[10];
    const int*   imw   = (const int*)d_in[11];
    float* out = (float*)d_out;

    // ---- carve workspace ----
    char* p = (char*)d_ws;
    size_t off = 0;
    auto carve = [&](size_t bytes) { void* q = p + off; off += align256(bytes); return q; };
    __bf16* A6   = (__bf16*)carve((size_t)MPAD * DK * 2);          // 51.4 MB
    __bf16* W6b  = (__bf16*)carve((size_t)F6 * DK * 2);            // 51.4 MB
    __bf16* W7b  = (__bf16*)carve((size_t)F6 * F6 * 2);
    __bf16* Wcb  = (__bf16*)carve((size_t)NCLS_PAD * F6 * 2);
    __bf16* Wbb  = (__bf16*)carve((size_t)NBOX * F6 * 2);
    __bf16* h6b  = (__bf16*)carve((size_t)MPAD * F6 * 2);
    __bf16* h7b  = (__bf16*)carve((size_t)MPAD * F6 * 2);
    float*  clsf = (float*)carve((size_t)MPAD * NCLS_PAD * 4);
    float*  bbxf = (float*)carve((size_t)MPAD * NBOX * 4);
    float*  boxes  = (float*)carve((size_t)NEL * 4 * 4);
    float*  raw    = (float*)carve((size_t)NEL * 4);
    float*  masked = (float*)carve((size_t)NEL * 4);
    int*    valid  = (int*)  carve((size_t)NEL * 4);
    float*  top_b  = (float*)carve((size_t)TOPK * 4 * 4);
    float*  top_s  = (float*)carve((size_t)TOPK * 4);
    int*    top_l  = (int*)  carve((size_t)TOPK * 4);
    int*    top_v  = (int*)  carve((size_t)TOPK * 4);

    // ---- 1. weight conversion (once per call) ----
    {
        long long n;
        n = (long long)F6 * DK;
        cvt_pad_bf16<<<(unsigned)((n + 255) / 256), 256, 0, stream>>>(W6, W6b, F6, DK, F6);
        n = (long long)F6 * F6;
        cvt_pad_bf16<<<(unsigned)((n + 255) / 256), 256, 0, stream>>>(W7, W7b, F6, F6, F6);
        n = (long long)NCLS_PAD * F6;
        cvt_pad_bf16<<<(unsigned)((n + 255) / 256), 256, 0, stream>>>(Wc, Wcb, NCLS, F6, NCLS_PAD);
        n = (long long)NBOX * F6;
        cvt_pad_bf16<<<(unsigned)((n + 255) / 256), 256, 0, stream>>>(Wb, Wbb, NBOX, F6, NBOX);
    }

    // ---- 2. ROI pool into bf16 A matrix ----
    roi_pool_kernel<<<MPAD, 256, 0, stream>>>(feat, props, A6);

    // ---- 3-6. WMMA GEMM chain (2x2 register-blocked macro-tiles) ----
    {
        // FC6: [1024 x 25088] x [25088 -> 1024], relu, bf16 out
        int tiles = (MPAD / 32) * (F6 / 32);
        gemm_wmma_bf16<2,2><<<(tiles + 3) / 4, 128, 0, stream>>>(
            A6, DK, W6b, DK, b6, F6, nullptr, h6b, F6, MPAD, F6, DK, 1);
        // FC7: [1024 x 1024] x [1024 -> 1024], relu, bf16 out
        gemm_wmma_bf16<2,2><<<(tiles + 3) / 4, 128, 0, stream>>>(
            h6b, F6, W7b, F6, b7, F6, nullptr, h7b, F6, MPAD, F6, F6, 1);
        // cls head: [1024 x 1024] x [1024 -> 64], f32 out
        int tc = (MPAD / 32) * (NCLS_PAD / 32);
        gemm_wmma_bf16<2,2><<<(tc + 3) / 4, 128, 0, stream>>>(
            h7b, F6, Wcb, F6, bc, NCLS, clsf, nullptr, NCLS_PAD, MPAD, NCLS_PAD, F6, 0);
        // bbox head: [1024 x 1024] x [1024 -> 208], f32 out (N=208 -> 2x1 tiles)
        int tb = (MPAD / 32) * (NBOX / 16);
        gemm_wmma_bf16<2,1><<<(tb + 3) / 4, 128, 0, stream>>>(
            h7b, F6, Wbb, F6, bb, NBOX, bbxf, nullptr, NBOX, MPAD, NBOX, F6, 0);
    }

    // ---- 7. decode ----
    decode_kernel<<<(NPROP + 127) / 128, 128, 0, stream>>>(clsf, bbxf, props, imh, imw,
                                                           boxes, raw, masked, valid);
    // ---- 8. top-2000 ----
    topk_rank_kernel<<<(NEL + 255) / 256, 256, 0, stream>>>(masked, raw, boxes, valid,
                                                            top_b, top_s, top_l, top_v);
    // ---- 9. NMS + scatter ----
    nms_scatter_kernel<<<1, 1024, 0, stream>>>(top_b, top_s, top_l, top_v, out);
}
